// MultiHeadAttentionBlock_15513421873623
// MI455X (gfx1250) — matrix operand order repaired and checked by reading, not compiled, not run on hardware
//
#include <hip/hip_runtime.h>
#include <hip/hip_bf16.h>

typedef __attribute__((ext_vector_type(16))) __bf16       v16bf;
typedef __attribute__((ext_vector_type(8)))  float        v8f;
typedef __attribute__((ext_vector_type(4)))  unsigned int u32x4;
typedef __attribute__((ext_vector_type(8)))  int          i32x8;
typedef __attribute__((ext_vector_type(4)))  int          i32x4;

#define HID   4096
#define HEADS 32
#define DK    128
#define SEQ   2048
#define BATCH 2

#if defined(__has_builtin)
#  if __has_builtin(__builtin_amdgcn_tensor_load_to_lds)
#    define HAVE_TDM 1
#  else
#    define HAVE_TDM 0
#  endif
#else
#  define HAVE_TDM 0
#endif

#if HAVE_TDM
// TDM load of a V chunk: 32 keys x 128 dims bf16 -> LDS in B-operand order.
// Tile scan order (X=16 dims contig, Y=32 keys stride 4096, Z=8 chunks stride 16)
// fills LDS linearly == vbuf[z*512 + y*16 + x]. Descriptor per ISA 8.3-8.6.
__device__ __forceinline__ void tdm_load_v_chunk(const __bf16* gsrc, __bf16* lds_dst)
{
    unsigned long long ga = (unsigned long long)(uintptr_t)gsrc;
    unsigned int       la = (unsigned int)(uintptr_t)(void*)lds_dst;

    u32x4 g0;
    g0[0] = 1u;                                          // count=1 (valid, user)
    g0[1] = la;                                          // lds_addr (bytes)
    g0[2] = (unsigned int)(ga & 0xFFFFFFFFu);            // global_addr[31:0]
    g0[3] = (unsigned int)((ga >> 32) & 0x01FFFFFFu)     // global_addr[56:32]
          | (2u << 30);                                  // type=2 ("image")

    i32x8 g1;
    g1[0] = (1 << 16);          // workgroup_mask=0, data_size=1 (2 bytes)
    g1[1] = (4096 << 16);       // tensor_dim0[15:0]=4096
    g1[2] = (2048 << 16);       // tensor_dim0 hi=0, tensor_dim1[15:0]=2048
    g1[3] = (16 << 16);         // tensor_dim1 hi=0, tile_dim0=16
    g1[4] = 32 | (8 << 16);     // tile_dim1=32 keys, tile_dim2=8 chunks
    g1[5] = 4096;               // tensor_dim0_stride lo32 (elements)
    g1[6] = (16 << 16);        // dim0_stride hi=0, tensor_dim1_stride lo16=16
    g1[7] = 0;                  // dim1_stride hi=0

    i32x4 g2;
    g2[0] = 8;                  // tensor_dim2 = 8
    g2[1] = 0;                  // tensor_dim3 (unused)
    g2[2] = 0;                  // tensor_dim2_stride (unused)
    g2[3] = 0;                  // tile_dim3 = 0
    i32x4 g3 = {0, 0, 0, 0};

#if defined(__clang_major__) && __clang_major__ >= 23
    i32x8 g4 = {0, 0, 0, 0, 0, 0, 0, 0};
    __builtin_amdgcn_tensor_load_to_lds(g0, g1, g2, g3, g4, 0);
#else
    __builtin_amdgcn_tensor_load_to_lds(g0, g1, g2, g3, 0);
#endif
}

__device__ __forceinline__ void wait_tensorcnt0()
{
#if __has_builtin(__builtin_amdgcn_s_wait_tensorcnt)
    __builtin_amdgcn_s_wait_tensorcnt(0);
#else
    asm volatile("s_wait_tensorcnt 0" ::: "memory");
#endif
}
#endif // HAVE_TDM

// ---------------------------------------------------------------------------
// GEMM:  C[M, HID] = A[M, HID] * W^T   (W [HID, HID] row-major f32)
// Block = 256 threads (8 waves), 64(M) x 128(N) tile, K-step 32.
// Each wave: one A operand reused across 4 B operands -> 4 WMMAs / K-step.
// LDS tiles pre-swizzled into WMMA operand layouts (ISA 7.12.2):
//   A op (16x32): lane L = row m=L%16; elem i -> k=(i&7)+16*(i>>3)+8*(L>>4)
//   B op (32x16): lane L = col n=L%16; elem i -> k=(i&7)+16*(i>>3)+8*(L>>4)
// ---------------------------------------------------------------------------
template<bool A_IS_BF16, bool OUT_BF16>
__global__ __launch_bounds__(256)
void gemm_wmma(const void* __restrict__ Ap, const float* __restrict__ W,
               void* __restrict__ Cp)
{
    __shared__ __align__(32) __bf16 abuf[4 * 512];   // 4 m-groups x (32 lanes x 16)
    __shared__ __align__(32) __bf16 bbuf[8 * 512];   // 8 n-groups

    const int tid  = threadIdx.x;
    const int lane = tid & 31;
    const int wave = tid >> 5;
    const int mi   = wave & 3;          // m-group 0..3
    const int nh   = wave >> 2;         // n half 0..1 (4 operands each)
    const int rowBase = blockIdx.y * 64;
    const int colBase = blockIdx.x * 128;

    const float*  Af = (const float*)Ap;
    const __bf16* Ab = (const __bf16*)Ap;

    v8f acc[4];
#pragma unroll
    for (int t = 0; t < 4; ++t) acc[t] = (v8f){0.f,0.f,0.f,0.f,0.f,0.f,0.f,0.f};

    for (int kb = 0; kb < HID; kb += 32) {
        // ---- stage A tile (64 rows x 32 k) -> bf16, A-operand order (8/thread)
#pragma unroll
        for (int j = 0; j < 8; ++j) {
            int p = tid * 8 + j;
            int g = p >> 9;
            int l = (p >> 4) & 31;
            int i = p & 15;
            int m_local = g * 16 + (l & 15);
            int k_local = (i & 7) + 16 * (i >> 3) + 8 * (l >> 4);
            size_t src = (size_t)(rowBase + m_local) * HID + kb + k_local;
            float v = A_IS_BF16 ? (float)Ab[src] : Af[src];
            abuf[p] = (__bf16)v;
        }
        // ---- stage B tile (128 n x 32 k) from W, B-operand order (16/thread)
#pragma unroll
        for (int j = 0; j < 16; ++j) {
            int p = tid * 16 + j;
            int g = p >> 9;                 // n-group
            int l = (p >> 4) & 31;          // operand lane: n = l%16
            int i = p & 15;                 // elem: k = (i&7)+16*(i>>3)+8*(l>>4)
            size_t src = (size_t)(colBase + g * 16 + (l & 15)) * HID + kb + (i & 7) + 16 * (i >> 3) + 8 * (l >> 4);
            bbuf[p] = (__bf16)W[src];
        }
        if (kb + 32 < HID) {                // -> global_prefetch_b8
            __builtin_prefetch(&W[(size_t)(colBase + (tid >> 1)) * HID + kb + 32], 0, 1);
            if (!A_IS_BF16)
                __builtin_prefetch(&Af[(size_t)(rowBase + (tid >> 2)) * HID + kb + 32], 0, 1);
        }
        __syncthreads();

        v16bf a = *(const v16bf*)&abuf[mi * 512 + lane * 16];
#pragma unroll
        for (int t = 0; t < 4; ++t) {
            v16bf b = *(const v16bf*)&bbuf[(nh * 4 + t) * 512 + lane * 16];
            acc[t] = __builtin_amdgcn_wmma_f32_16x16x32_bf16(false, a, false, b,
                                                             (short)0, acc[t], false, false);
        }
        __syncthreads();
    }

    // ---- epilogue: C element (g, lane): row = g + 8*(lane>=16), col = lane%16
    const int row0 = rowBase + mi * 16 + 8 * (lane >> 4);
    const int col0 = colBase + nh * 64 + (lane & 15);
#pragma unroll
    for (int t = 0; t < 4; ++t) {
#pragma unroll
        for (int g = 0; g < 8; ++g) {
            size_t o = (size_t)(row0 + g) * HID + col0 + t * 16;
            if (OUT_BF16) ((__bf16*)Cp)[o] = (__bf16)acc[t][g];
            else          ((float*)Cp)[o]  = acc[t][g];
        }
    }
}

// ---------------------------------------------------------------------------
// Flash attention: block = 256 threads (8 waves), one (b, h, 128-query) slab.
// Each wave owns 16 query rows. Key loop processes 32 keys per step.
// V chunk staged by the Tensor Data Mover (TDM) when available.
// ---------------------------------------------------------------------------
__global__ __launch_bounds__(256)
void attn_wmma(const __bf16* __restrict__ Q, const __bf16* __restrict__ K,
               const __bf16* __restrict__ V, __bf16* __restrict__ O)
{
    __shared__ __align__(32) __bf16 qbuf[8 * 4 * 512];  // per-wave Q A-operands
    __shared__ __align__(32) __bf16 kbuf[8 * 512];      // K B-ops (2 keySub x 4 dim)
    __shared__ __align__(32) __bf16 vbuf[8 * 512];      // V B-ops (8 dim chunks)
    __shared__ __align__(32) __bf16 pbuf[8 * 512];      // per-wave P A-operand

    const int tid  = threadIdx.x;
    const int lane = tid & 31;
    const int wave = tid >> 5;

    int idx = blockIdx.x;
    const int qt = idx & 15;  idx >>= 4;    // SEQ/128 = 16 query slabs
    const int h  = idx & 31;  idx >>= 5;    // HEADS
    const int b  = idx;                     // BATCH
    const int    qBase = qt * 128;
    const size_t hOff  = (size_t)h * DK;
    const size_t bOff  = (size_t)b * SEQ * HID;

    // ---- stage Q (128 rows x 128 dims) into A-operand layout
    for (int j = 0; j < 64; ++j) {
        int p = tid * 64 + j;
        int w = p >> 11;
        int c = (p >> 9) & 3;
        int l = (p >> 4) & 31;
        int i = p & 15;
        int m = l & 15;
        int k = (i & 7) + 16 * (i >> 3) + 8 * (l >> 4);
        qbuf[p] = Q[bOff + (size_t)(qBase + w * 16 + m) * HID + hOff + c * 32 + k];
    }
    __syncthreads();

    v16bf qa[4];
#pragma unroll
    for (int c = 0; c < 4; ++c)
        qa[c] = *(const v16bf*)&qbuf[wave * 2048 + c * 512 + lane * 16];

    float m_run[8], l_run[8];
    v8f   oacc[8];
#pragma unroll
    for (int g = 0; g < 8; ++g) { m_run[g] = -1e30f; l_run[g] = 0.f; }
#pragma unroll
    for (int j = 0; j < 8; ++j) oacc[j] = (v8f){0.f,0.f,0.f,0.f,0.f,0.f,0.f,0.f};

    const float scale = 0.08838834764831845f;   // 1/sqrt(128)

    for (int kt = 0; kt < SEQ; kt += 32) {
#if HAVE_TDM
        if (wave == 0)
            tdm_load_v_chunk(&V[bOff + (size_t)kt * HID + hOff], vbuf);
#endif
        // ---- cooperative staging of K chunk (32 keys x 128 dims), transposed
        {
            const int r  = tid >> 3;            // key 0..31 within chunk
            const int d0 = (tid & 7) * 16;      // dim base
            const __bf16* ksrc = &K[bOff + (size_t)(kt + r) * HID + hOff + d0];
#pragma unroll
            for (int j = 0; j < 16; ++j) {
                int d = d0 + j;
                // K B-operand (32 dim x 16 key): op = keySub*4 + dimChunk
                kbuf[((r >> 4) * 4 + (d >> 5)) * 512 + ((r & 15) + 16 * ((d >> 3) & 1)) * 16 + (d & 7) + 8 * ((d & 31) >> 4)] = ksrc[j];
            }
#if !HAVE_TDM
            const __bf16* vsrc = &V[bOff + (size_t)(kt + r) * HID + hOff + d0];
#pragma unroll
            for (int j = 0; j < 16; ++j) {
                int d = d0 + j;
                vbuf[(d >> 4) * 512 + r * 16 + (d & 15)] = vsrc[j];
            }
#endif
        }
#if HAVE_TDM
        if (wave == 0) wait_tensorcnt0();
#endif
        __syncthreads();

        // ---- scores for two 16-key subtiles: S = Q(16x128) . K^T
        v8f s0 = {0.f,0.f,0.f,0.f,0.f,0.f,0.f,0.f};
        v8f s1 = {0.f,0.f,0.f,0.f,0.f,0.f,0.f,0.f};
#pragma unroll
        for (int dj = 0; dj < 4; ++dj) {
            v16bf kb0 = *(const v16bf*)&kbuf[(0 * 4 + dj) * 512 + lane * 16];
            v16bf kb1 = *(const v16bf*)&kbuf[(1 * 4 + dj) * 512 + lane * 16];
            s0 = __builtin_amdgcn_wmma_f32_16x16x32_bf16(false, qa[dj], false, kb0,
                                                         (short)0, s0, false, false);
            s1 = __builtin_amdgcn_wmma_f32_16x16x32_bf16(false, qa[dj], false, kb1,
                                                         (short)0, s1, false, false);
        }

        // ---- online softmax. C layout: lane holds key n=lane%16, row g+8*(lane>>4)
        const int kl     = lane & 15;
        const int laneHi = (kl >> 3) & 1;
#pragma unroll
        for (int g = 0; g < 8; ++g) {
            float v0 = s0[g] * scale;
            float v1 = s1[g] * scale;
            float mx = fmaxf(v0, v1);
#pragma unroll
            for (int x = 8; x >= 1; x >>= 1)
                mx = fmaxf(mx, __shfl_xor(mx, x, 32));      // 16-lane half reduce
            float m_new = fmaxf(m_run[g], mx);
            float alpha = __expf(m_run[g] - m_new);
            float p0 = __expf(v0 - m_new);
            float p1 = __expf(v1 - m_new);
            float rs = p0 + p1;
#pragma unroll
            for (int x = 8; x >= 1; x >>= 1)
                rs += __shfl_xor(rs, x, 32);
            l_run[g] = l_run[g] * alpha + rs;
            m_run[g] = m_new;
#pragma unroll
            for (int j = 0; j < 8; ++j) oacc[j][g] *= alpha;

            // transpose P into A-operand layout via per-wave LDS region
            int la = (g + 8 * (lane >> 4)) + 16 * laneHi;
            pbuf[wave * 512 + la * 16 + (kl & 7)    ] = (__bf16)p0;
            pbuf[wave * 512 + la * 16 + (kl & 7) + 8] = (__bf16)p1;
        }

        // same-wave DS ordering makes the stores above visible to this load
        v16bf pa = *(const v16bf*)&pbuf[wave * 512 + lane * 16];
#pragma unroll
        for (int j = 0; j < 8; ++j) {
            v16bf vb;
#pragma unroll
            for (int s = 0; s < 16; ++s)
                vb[s] = vbuf[j * 512 + ((s & 7) + 8 * (lane >> 4) + 16 * (s >> 3)) * 16 + (lane & 15)];
            oacc[j] = __builtin_amdgcn_wmma_f32_16x16x32_bf16(false, pa, false, vb,
                                                              (short)0, oacc[j], false, false);
        }
        __syncthreads();
    }

    // ---- normalize and store (bf16, [b, s, 4096] layout)
#pragma unroll
    for (int j = 0; j < 8; ++j) {
#pragma unroll
        for (int g = 0; g < 8; ++g) {
            int row = qBase + wave * 16 + g + 8 * (lane >> 4);
            int d   = j * 16 + (lane & 15);
            float val = oacc[j][g] / l_run[g];
            O[bOff + (size_t)row * HID + hOff + d] = (__bf16)val;
        }
    }
}

// ---------------------------------------------------------------------------
extern "C" void kernel_launch(void* const* d_in, const int* in_sizes, int n_in,
                              void* d_out, int out_size, void* d_ws, size_t ws_size,
                              hipStream_t stream)
{
    const float* x   = (const float*)d_in[0];
    const float* w_q = (const float*)d_in[1];
    const float* w_k = (const float*)d_in[2];
    const float* w_v = (const float*)d_in[3];
    const float* w_o = (const float*)d_in[4];
    float* out = (float*)d_out;

    const size_t NTOK = (size_t)BATCH * SEQ;          // 4096 rows
    const size_t BUF  = NTOK * HID;                   // 16,777,216 elements

    __bf16* Qb = (__bf16*)d_ws;
    __bf16* Kb = Qb + BUF;
    __bf16* Vb = Kb + BUF;
    __bf16* Ob = Vb + BUF;

    dim3 grid(HID / 128, NTOK / 64);   // 64(M) x 128(N) tiles
    dim3 blk(256);

    gemm_wmma<false, true><<<grid, blk, 0, stream>>>((const void*)x, w_q, (void*)Qb);
    gemm_wmma<false, true><<<grid, blk, 0, stream>>>((const void*)x, w_k, (void*)Kb);
    gemm_wmma<false, true><<<grid, blk, 0, stream>>>((const void*)x, w_v, (void*)Vb);

    attn_wmma<<<BATCH * HEADS * (SEQ / 128), blk, 0, stream>>>(Qb, Kb, Vb, Ob);

    gemm_wmma<true, false><<<grid, blk, 0, stream>>>((const void*)Ob, w_o, (void*)out);
}
